// BigruCrf_24558622998985
// MI455X (gfx1250) — matrix operand order repaired and checked by reading, not compiled, not run on hardware
//
#include <hip/hip_runtime.h>
#include <hip/hip_bf16.h>
#include <math.h>

// ---------------------------------------------------------------------------
// BiGRU + CRF Viterbi for MI455X (gfx1250), wave32, WMMA f16->f32.
//
//   K1 pack_w   : f32 weights [300][100] + bias -> f16 padded [304][128],
//                 bias folded into column K=100 (augmented-ones trick)
//   K2 gru_wmma : 16 persistent blocks (2 dirs x 8 batch-tiles of 16 rows).
//                 Weight B-fragments live in VGPRs for all 2048 steps;
//                 x tile staged to LDS once per step; next-step x prefetched
//                 via global_prefetch_b8.
//   K3 feats    : [T][31] projection of concat(fwd,bwd) for `seq`
//   K4 viterbi  : single wave32, backpointers in LDS, backtrack on lane 0
// ---------------------------------------------------------------------------

typedef __attribute__((ext_vector_type(16))) _Float16 v16h;
typedef __attribute__((ext_vector_type(8)))  float    v8f;

#define HID   100
#define GAT   300     // 3*HID, PyTorch gate order [r, z, n]
#define GP    304     // gates padded to multiple of 16
#define KP    128     // K (=100 features + 1 bias) padded to multiple of 32
#define BATCH 128
#define NTAG  31
#define NEGV  (-10000.0f)
#define TMAX  2048

// ---------------------------------------------------------------- K1: pack
// dst[w][n][k]: k<100 -> W[n][k]; k==100 -> bias[n]; else 0
__global__ void bigru_pack_w(const float* __restrict__ wa, const float* __restrict__ ba,
                             const float* __restrict__ wb, const float* __restrict__ bb,
                             const float* __restrict__ wc, const float* __restrict__ bc,
                             const float* __restrict__ wd, const float* __restrict__ bd,
                             _Float16* __restrict__ dst) {
    int n = blockIdx.x;            // 0..303 (gate row)
    int k = threadIdx.x;           // 0..127
    int w = blockIdx.y;            // 0..3  : (wih,bih)_f, (whh,bhh)_f, (wih,bih)_b, (whh,bhh)_b
    const float* W  = (w == 0) ? wa : (w == 1) ? wb : (w == 2) ? wc : wd;
    const float* Bv = (w == 0) ? ba : (w == 1) ? bb : (w == 2) ? bc : bd;
    float v = 0.0f;
    if (n < GAT) {
        if (k < HID)       v = W[n * HID + k];
        else if (k == HID) v = Bv[n];
    }
    dst[((size_t)w * GP + n) * KP + k] = (_Float16)v;
}

// ---------------------------------------------------------------- K2: GRU
// blockIdx.x: [0..15] -> dir = x>>3 (0 fwd, 1 bwd), bt = x&7 (batch tile of 16)
__global__ __launch_bounds__(256)
__attribute__((amdgpu_waves_per_eu(1)))
void bigru_gru_wmma(const float* __restrict__ x,        // [T][128][100] f32
                    const _Float16* __restrict__ w16,   // 4 x [304][128] f16 (bias folded)
                    const int* __restrict__ seqp,
                    float* __restrict__ hseq,           // [2][T][128] f32 (only seq row used)
                    int T) {
    const int tid  = threadIdx.x;
    const int wave = tid >> 5;
    const int lane = tid & 31;
    const int dir  = blockIdx.x >> 3;
    const int bt   = blockIdx.x & 7;
    const int b0   = bt * 16;

    const _Float16* wih = w16 + (size_t)(dir * 2)     * GP * KP;
    const _Float16* whh = w16 + (size_t)(dir * 2 + 1) * GP * KP;
    float* hout = hseq + (size_t)dir * T * BATCH;

    const int seqv  = *seqp;
    const bool owns = (seqv >= b0) && (seqv < b0 + 16);
    const int  mrow = seqv - b0;

    __shared__ float    gi_s[16][GP];     // x @ [W_ih^T; b_ih]  (f32 accum, bias included)
    __shared__ float    gh_s[16][GP];     // h @ [W_hh^T; b_hh]
    __shared__ float    h32[16][104];     // hidden state, f32 master copy
    __shared__ _Float16 h16[16][KP];      // hidden state, f16 operand; col 100 == 1.0 (bias)
    __shared__ _Float16 xs [16][KP];      // staged x tile,  f16 operand; col 100 == 1.0

    for (int i = tid; i < 16 * KP; i += 256) {
        const int r = i >> 7, k = i & (KP - 1);
        const _Float16 v = (k == HID) ? (_Float16)1.0f : (_Float16)0.0f;
        h16[r][k] = v;
        xs [r][k] = v;      // cols >= 100 are never rewritten below
    }
    for (int i = tid; i < 16 * 104; i += 256) h32[i / 104][i % 104] = 0.0f;

    const int Arow = lane & 15;           // A-matrix M (lanes 0-15 and 16-31 both M=0..15)
    const int half = lane >> 4;           // lane half selects K sub-range

    // ---- hoist B-fragments (weights) into persistent VGPRs -----------------
    // slot sl owns N-tile j = wave + 8*sl (19 tiles of 16 cols over 8 waves)
    v16h bi_frag[3][4], bh_frag[3][4];
    int  tileN[3];
    bool tvalid[3];
#pragma unroll
    for (int sl = 0; sl < 3; ++sl) {
        const int j = wave + sl * 8;
        tvalid[sl] = (j < GP / 16);
        const int jj = tvalid[sl] ? j : 0;
        const int N = jj * 16 + (lane & 15);
        tileN[sl] = N;
#pragma unroll
        for (int c = 0; c < 4; ++c) {
#pragma unroll
            for (int p = 0; p < 8; ++p) {
                const int K = c * 32 + half * 16 + 2 * p;   // B 32x16 layout
                bi_frag[sl][c][2 * p]     = wih[(size_t)N * KP + K];
                bi_frag[sl][c][2 * p + 1] = wih[(size_t)N * KP + K + 1];
                bh_frag[sl][c][2 * p]     = whh[(size_t)N * KP + K];
                bh_frag[sl][c][2 * p + 1] = whh[(size_t)N * KP + K + 1];
            }
        }
    }
    __syncthreads();

    for (int s = 0; s < T; ++s) {
        const int t = dir ? (T - 1 - s) : s;   // backward dir consumes reversed time

        // ---- stage x tile (16 rows x 100 f32, contiguous 6400 B) into LDS
        const float* xtile = x + ((size_t)t * BATCH + b0) * HID;
        for (int i = tid; i < 16 * HID; i += 256)
            xs[i / HID][i % HID] = (_Float16)xtile[i];
        __syncthreads();

        // ---- per-wave GEMM tiles, all operands in LDS/VGPRs
#pragma unroll
        for (int sl = 0; sl < 3; ++sl) {
            if (!tvalid[sl]) continue;          // wave-uniform
            v8f ci = {};
            v8f ch = {};
#pragma unroll
            for (int c = 0; c < 4; ++c) {
                v16h ax, ah;                    // A 16x32 f16 fragments from LDS
#pragma unroll
                for (int p = 0; p < 8; ++p) {
                    const int K = c * 32 + half * 16 + 2 * p;
                    ax[2 * p]     = xs [Arow][K];
                    ax[2 * p + 1] = xs [Arow][K + 1];
                    ah[2 * p]     = h16[Arow][K];
                    ah[2 * p + 1] = h16[Arow][K + 1];
                }
                ci = __builtin_amdgcn_wmma_f32_16x16x32_f16(false, ax, false, bi_frag[sl][c],
                                                            (short)0, ci, false, false);
                ch = __builtin_amdgcn_wmma_f32_16x16x32_f16(false, ah, false, bh_frag[sl][c],
                                                            (short)0, ch, false, false);
            }
            const int N = tileN[sl];
#pragma unroll
            for (int v = 0; v < 8; ++v) {       // C/D: VGPR v -> M = v + 8*half
                const int M = v + half * 8;
                gi_s[M][N] = ci[v];
                gh_s[M][N] = ch[v];
            }
        }
        __syncthreads();

        // ---- prefetch next step's x tile (50 x 128B cachelines)
        if (s + 1 < T) {
            const int tn = dir ? (T - 2 - s) : (s + 1);
            const char* pf = (const char*)(x + ((size_t)tn * BATCH + b0) * HID);
            if (tid < 50) __builtin_prefetch(pf + (size_t)tid * 128, 0, 1);
        }

        // ---- gates + state update: 16x100 outputs over 256 threads
        for (int i = tid; i < 16 * HID; i += 256) {
            const int m = i / HID, jj = i % HID;
            const float r = 1.0f / (1.0f + __expf(-(gi_s[m][jj]       + gh_s[m][jj])));
            const float z = 1.0f / (1.0f + __expf(-(gi_s[m][HID + jj] + gh_s[m][HID + jj])));
            const float g = tanhf(gi_s[m][2 * HID + jj] + r * gh_s[m][2 * HID + jj]);
            const float hnew = (1.0f - z) * g + z * h32[m][jj];
            h32[m][jj] = hnew;
            h16[m][jj] = (_Float16)hnew;
            if (owns && m == mrow) hout[(size_t)t * BATCH + jj] = hnew;
        }
        __syncthreads();
    }
}

// ---------------------------------------------------------------- K3: feats
__global__ __launch_bounds__(32)
void bigru_feats(const float* __restrict__ fwd, const float* __restrict__ bwd,
                 const float* __restrict__ w_out, const float* __restrict__ b_out,
                 float* __restrict__ feats, int T) {
    const int t = blockIdx.x;
    const int n = threadIdx.x;
    if (n < NTAG) {
        float acc = b_out[n];
        const float* wr = w_out + n * (2 * HID);
        const float* f  = fwd + (size_t)t * BATCH;
        const float* b  = bwd + (size_t)t * BATCH;
        for (int j = 0; j < HID; ++j)
            acc += f[j] * wr[j] + b[j] * wr[HID + j];
        feats[(size_t)t * 32 + n] = acc;
    }
}

// ---------------------------------------------------------------- K4: viterbi
__global__ __launch_bounds__(32)
void bigru_viterbi(const float* __restrict__ feats, const float* __restrict__ mask,
                   const float* __restrict__ trans, const int* __restrict__ seqp,
                   float* __restrict__ out, int T) {
    __shared__ float fv[32];
    __shared__ unsigned char bp[TMAX][NTAG];   // 63488 B backpointers in LDS
    const int lane = threadIdx.x;
    const int seqv = *seqp;

    float tr[NTAG];
#pragma unroll
    for (int p = 0; p < NTAG; ++p)
        tr[p] = (lane < NTAG) ? trans[lane * NTAG + p] : NEGV;

    fv[lane] = (lane == 0) ? 0.0f : NEGV;
    __syncthreads();

    const int Tc = (T < TMAX) ? T : TMAX;
    for (int t = 0; t < Tc; ++t) {
        const float fvold = fv[lane];
        float best = -3.4e38f;
        int bidx = 0;
#pragma unroll
        for (int p = 0; p < NTAG; ++p) {
            const float sc = fv[p] + tr[p];     // scores[next=lane][prev=p]
            if (sc > best) { best = sc; bidx = p; }
        }
        const float feat = feats[(size_t)t * 32 + lane];
        const float m = mask[(size_t)seqv * T + t];
        const bool live = (m > 0.0f);
        const float vv = best + feat;
        __syncthreads();                         // all reads of fv done
        fv[lane] = live ? vv : fvold;
        if (lane < NTAG)
            bp[t][lane] = (unsigned char)(live ? bidx : lane);
        __syncthreads();
    }

    if (lane == 0) {
        int besttag = 0;
        float bs = fv[0];
        for (int i = 1; i < NTAG; ++i)
            if (fv[i] > bs) { bs = fv[i]; besttag = i; }
        out[0] = bs;
        int tag = besttag;
        for (int t = Tc - 1; t >= 0; --t) {
            out[1 + t] = (float)tag;             // path[t]
            tag = bp[t][tag];
        }
    }
}

// ---------------------------------------------------------------- launch
extern "C" void kernel_launch(void* const* d_in, const int* in_sizes, int n_in,
                              void* d_out, int out_size, void* d_ws, size_t ws_size,
                              hipStream_t stream) {
    const float* x      = (const float*)d_in[0];   // [T,128,100]
    const float* mask   = (const float*)d_in[1];   // [128,T]
    const float* w_ih_f = (const float*)d_in[2];
    const float* w_hh_f = (const float*)d_in[3];
    const float* b_ih_f = (const float*)d_in[4];
    const float* b_hh_f = (const float*)d_in[5];
    const float* w_ih_b = (const float*)d_in[6];
    const float* w_hh_b = (const float*)d_in[7];
    const float* b_ih_b = (const float*)d_in[8];
    const float* b_hh_b = (const float*)d_in[9];
    const float* w_out  = (const float*)d_in[10];  // [31,200]
    const float* b_out  = (const float*)d_in[11];  // [31]
    const float* trans  = (const float*)d_in[12];  // [31,31]
    const int*   seqp   = (const int*)d_in[14];

    const int T = in_sizes[1] / BATCH;             // 2048

    // workspace layout (floats then f16):
    float* fwd   = (float*)d_ws;                   // [T][128]
    float* bwd   = fwd + (size_t)T * BATCH;        // [T][128]
    float* feats = bwd + (size_t)T * BATCH;        // [T][32]
    _Float16* w16 = (_Float16*)(feats + (size_t)T * 32);  // 4 x [304][128]

    // K1: pack the 4 weight matrices (+ folded biases) to padded f16
    bigru_pack_w<<<dim3(GP, 4), KP, 0, stream>>>(w_ih_f, b_ih_f, w_hh_f, b_hh_f,
                                                 w_ih_b, b_ih_b, w_hh_b, b_hh_b, w16);

    // K2: persistent bidirectional GRU (16 blocks = 2 dirs x 8 batch tiles)
    bigru_gru_wmma<<<16, 256, 0, stream>>>(x, w16, seqp, fwd, T);

    // K3: tag projection for the selected sequence
    bigru_feats<<<T, 32, 0, stream>>>(fwd, bwd, w_out, b_out, feats, T);

    // K4: Viterbi decode (score + path) -> d_out[0], d_out[1..T]
    bigru_viterbi<<<1, 32, 0, stream>>>(feats, mask, trans, seqp, (float*)d_out, T);
}
